// PointNet2cls_19000935317876
// MI455X (gfx1250) — matrix operand order repaired and checked by reading, not compiled, not run on hardware
//
#include <hip/hip_runtime.h>

// ---------------------------------------------------------------------------
// PointNet++ classification for MI455X (gfx1250, wave32, WMMA + TDM).
// B=16, N=2048, HID=64, OUT=40.
// d_out = [logits (16*40)] ++ [l3p transposed (16*512)]  (f32, 8832 elems)
// ---------------------------------------------------------------------------

typedef __attribute__((ext_vector_type(16))) _Float16 v16h;
typedef __attribute__((ext_vector_type(8)))  float    v8f;
typedef unsigned int u32x4 __attribute__((ext_vector_type(4)));
typedef int          i32x8 __attribute__((ext_vector_type(8)));
typedef int          i32x4 __attribute__((ext_vector_type(4)));

union frag16 { v16h h; uint4 u[2]; };
union h8pack { uint4 u; _Float16 h[8]; };

#define EPS_BN 1e-5f

#if defined(__has_builtin)
#if __has_builtin(__builtin_amdgcn_tensor_load_to_lds)
#define USE_TDM 1
#endif
#endif
#ifndef USE_TDM
#define USE_TDM 0
#endif

#if USE_TDM
// Build a 2D Tensor-DMA descriptor (D# groups 0/1) and issue tensor_load_to_lds.
// Tile = tileX (contiguous halves) x tileY rows; row pitch strideElems halves.
// Tensor dims are set equal to the tile dims (tile is always fully in-bounds),
// the real layout is carried by tensor_dim0_stride.
// 6-arg builtin form (clang-23 / therock-10.0): groups 2/3 + trailing i32x8
// are unused for 2D tiles and passed as zeros.
__device__ __forceinline__ void tdm_load_2d(unsigned lds_addr, const _Float16* gptr,
                                            unsigned tileX, unsigned tileY,
                                            unsigned strideElems) {
    unsigned long long ga = (unsigned long long)(size_t)(const void*)gptr;
    unsigned long long s0 = (unsigned long long)strideElems;
    unsigned td0 = tileX, td1 = tileY;

    u32x4 g0;
    g0[0] = 1u;                                            // count=1, no gather
    g0[1] = lds_addr;                                      // LDS byte address
    g0[2] = (unsigned)(ga & 0xffffffffull);                // global_addr[31:0]
    g0[3] = (unsigned)((ga >> 32) & 0x01ffffffull)         // global_addr[56:32]
          | (2u << 30);                                    // type=2 ("image")

    i32x8 g1;
    g1[0] = (int)(1u << 16);                               // wg_mask=0, data_size=1 (2B)
    g1[1] = (int)((td0 & 0xffffu) << 16);                  // tensor_dim0[15:0]
    g1[2] = (int)(((td0 >> 16) & 0xffffu) | ((td1 & 0xffffu) << 16));
    g1[3] = (int)(((td1 >> 16) & 0xffffu) | ((tileX & 0xffffu) << 16)); // tile_dim0
    g1[4] = (int)(tileY & 0xffffu);                        // tile_dim1; tile_dim2=0
    g1[5] = (int)(unsigned)(s0 & 0xffffffffull);           // dim0_stride[31:0]
    g1[6] = (int)(unsigned)((s0 >> 32) & 0xffffull);       // dim0_stride[47:32]; dim1_stride=0
    g1[7] = 0;

    i32x4 z4; z4[0] = 0; z4[1] = 0; z4[2] = 0; z4[3] = 0;
    i32x8 z8; z8[0] = 0; z8[1] = 0; z8[2] = 0; z8[3] = 0;
    z8[4] = 0; z8[5] = 0; z8[6] = 0; z8[7] = 0;
    __builtin_amdgcn_tensor_load_to_lds(g0, g1, z4, z4, z8, 0);
}
#endif

// ------------------------------ transpose (B,3,N) -> (B,N,3) ---------------
__global__ void k_transpose_xyz(const float* __restrict__ src, float* __restrict__ dst) {
    int i = blockIdx.x * 256 + threadIdx.x;
    if (i >= 16 * 2048 * 3) return;
    int b = i / (2048 * 3);
    int rem = i - b * (2048 * 3);
    int n = rem / 3, c = rem - n * 3;
    dst[i] = src[(size_t)b * 3 * 2048 + (size_t)c * 2048 + n];
}

// ------------------------------ farthest point sampling --------------------
__global__ void k_fps(const float* __restrict__ xyz,  // [B][n][3]
                      float* __restrict__ outXyz,     // [B][npoint][3]
                      int n, int npoint) {
    __shared__ float dist[2048];
    __shared__ float redv[1024];
    __shared__ int   redi[1024];
    __shared__ float cent[3];
    int b = blockIdx.x, t = threadIdx.x;
    const float* base = xyz + (size_t)b * n * 3;
    for (int i = t; i < n; i += 1024) dist[i] = 1e10f;
    __syncthreads();
    int far = 0;
    for (int it = 0; it < npoint; ++it) {
        if (t == 0) {
            float cx = base[far * 3 + 0], cy = base[far * 3 + 1], cz = base[far * 3 + 2];
            cent[0] = cx; cent[1] = cy; cent[2] = cz;
            float* o = outXyz + ((size_t)b * npoint + it) * 3;
            o[0] = cx; o[1] = cy; o[2] = cz;
        }
        __syncthreads();
        float cx = cent[0], cy = cent[1], cz = cent[2];
        float bv = -1.0f; int bi = 0;
        for (int i = t; i < n; i += 1024) {
            float dx = base[i * 3 + 0] - cx;
            float dy = base[i * 3 + 1] - cy;
            float dz = base[i * 3 + 2] - cz;
            float d = dx * dx + dy * dy + dz * dz;
            float nd = fminf(dist[i], d);
            dist[i] = nd;
            if (nd > bv) { bv = nd; bi = i; }  // ascending i -> first-max kept
        }
        redv[t] = bv; redi[t] = bi;
        __syncthreads();
        for (int s = 512; s > 0; s >>= 1) {
            if (t < s) {
                if (redv[t + s] > redv[t]) { redv[t] = redv[t + s]; redi[t] = redi[t + s]; }
            }
            __syncthreads();
        }
        far = redi[0];
    }
}

// ------------------------------ ball query + grouping ----------------------
__global__ void k_group_sa1(const float* __restrict__ xyzT, const float* __restrict__ nxyz,
                            _Float16* __restrict__ X) {
    __shared__ int nb[32];
    __shared__ float cc[3];
    int bs = blockIdx.x;          // 0..16383
    int b = bs >> 10;
    int t = threadIdx.x;          // 0..31
    const float* pts = xyzT + (size_t)b * 2048 * 3;
    if (t == 0) {
        const float* c = nxyz + (size_t)bs * 3;
        float cx = c[0], cy = c[1], cz = c[2];
        cc[0] = cx; cc[1] = cy; cc[2] = cz;
        int cnt = 0;
        for (int j = 0; j < 2048 && cnt < 32; ++j) {
            float dx = pts[j * 3 + 0] - cx, dy = pts[j * 3 + 1] - cy, dz = pts[j * 3 + 2] - cz;
            if (dx * dx + dy * dy + dz * dz <= 0.04f) nb[cnt++] = j;
        }
        int f = (cnt > 0) ? nb[0] : 0;
        for (int k = cnt; k < 32; ++k) nb[k] = f;
    }
    __syncthreads();
    int j = nb[t];
    _Float16* dst = X + ((size_t)bs * 32 + t) * 32;
    dst[0] = (_Float16)(pts[j * 3 + 0] - cc[0]);
    dst[1] = (_Float16)(pts[j * 3 + 1] - cc[1]);
    dst[2] = (_Float16)(pts[j * 3 + 2] - cc[2]);
    for (int k = 3; k < 32; ++k) dst[k] = (_Float16)0.0f;
}

__global__ void k_group_sa2(const float* __restrict__ nxyz1, const float* __restrict__ nxyz2,
                            const _Float16* __restrict__ l1p, _Float16* __restrict__ X) {
    __shared__ int nb[64];
    __shared__ float cc[3];
    int bs = blockIdx.x;          // 0..8191
    int b = bs >> 9;
    int t = threadIdx.x;          // 0..63
    const float* pts = nxyz1 + (size_t)b * 1024 * 3;
    if (t == 0) {
        const float* c = nxyz2 + (size_t)bs * 3;
        float cx = c[0], cy = c[1], cz = c[2];
        cc[0] = cx; cc[1] = cy; cc[2] = cz;
        int cnt = 0;
        for (int j = 0; j < 1024 && cnt < 64; ++j) {
            float dx = pts[j * 3 + 0] - cx, dy = pts[j * 3 + 1] - cy, dz = pts[j * 3 + 2] - cz;
            if (dx * dx + dy * dy + dz * dz <= 0.16f) nb[cnt++] = j;
        }
        int f = (cnt > 0) ? nb[0] : 0;
        for (int k = cnt; k < 64; ++k) nb[k] = f;
    }
    __syncthreads();
    int j = nb[t];
    _Float16* dst = X + ((size_t)bs * 64 + t) * 160;
    dst[0] = (_Float16)(pts[j * 3 + 0] - cc[0]);
    dst[1] = (_Float16)(pts[j * 3 + 1] - cc[1]);
    dst[2] = (_Float16)(pts[j * 3 + 2] - cc[2]);
    const _Float16* src = l1p + ((size_t)b * 1024 + j) * 128;
    for (int c2 = 0; c2 < 128; ++c2) dst[3 + c2] = src[c2];
    for (int c2 = 131; c2 < 160; ++c2) dst[c2] = (_Float16)0.0f;
}

__global__ void k_group_sa3(const float* __restrict__ nxyz2, const _Float16* __restrict__ l2p,
                            _Float16* __restrict__ X) {
    int i = blockIdx.x * 256 + threadIdx.x;
    if (i >= 8192 * 288) return;
    int row = i / 288, c = i - row * 288;
    _Float16 v;
    if (c < 3)        v = (_Float16)nxyz2[(size_t)row * 3 + c];
    else if (c < 259) v = l2p[(size_t)row * 256 + (c - 3)];
    else              v = (_Float16)0.0f;
    X[i] = v;
}

// ------------------------------ weight packing ------------------------------
__global__ void k_pack_w(const float* __restrict__ W, _Float16* __restrict__ Wt,
                         int cin, int cout, int kpad) {
    int i = blockIdx.x * 256 + threadIdx.x;
    if (i >= cout * kpad) return;
    int n = i / kpad, k = i - n * kpad;
    Wt[i] = (k < cin) ? (_Float16)W[(size_t)k * cout + n] : (_Float16)0.0f;
}

__global__ void k_zero_stats(float* cs, float* cq) {
    int t = threadIdx.x;  // 512 threads
    cs[t] = 0.0f; cq[t] = 0.0f;
}

// ------------------------------ WMMA GEMM + bias + BN stats -----------------
// Y[M][N] = A[M][K] * Wt[N][K]^T + bias, fp16 in, f32 accum, fp16 out.
// Block tile 128x64, 8 waves, each wave 32x32 (2x2 v_wmma_f32_16x16x32_f16).
// Tiles staged into LDS by the Tensor Data Mover (double-buffered on
// TENSORcnt); epilogue accumulates per-column sum / sum-sq for batchnorm.
__global__ void k_wmma_gemm_bn(const _Float16* __restrict__ A, const _Float16* __restrict__ Wt,
                               const float* __restrict__ bias, _Float16* __restrict__ Y,
                               float* __restrict__ colsum, float* __restrict__ colsumsq,
                               int M, int N, int K) {
    __shared__ alignas(16) _Float16 As[2 * 128 * 32];
    __shared__ alignas(16) _Float16 Bs[2 * 64 * 32];
    __shared__ float s_sum[64];
    __shared__ float s_sq[64];

    const int t = threadIdx.x;
    const int lane = t & 31, wave = t >> 5;
    const int wm = wave >> 1, wn = wave & 1;
    const int mBase = blockIdx.y * 128, nBase = blockIdx.x * 64;

    if (t < 64) { s_sum[t] = 0.0f; s_sq[t] = 0.0f; }

    v8f acc00 = {}; v8f acc01 = {}; v8f acc10 = {}; v8f acc11 = {};

    const int mrow = lane & 15;
    const int kA = (lane < 16) ? 0 : 8;
    const int kB = (lane < 16) ? 0 : 8;

#if USE_TDM
    if (wave == 0) {
        tdm_load_2d((unsigned)(size_t)(void*)&As[0], A + (size_t)mBase * K, 32, 128, (unsigned)K);
        tdm_load_2d((unsigned)(size_t)(void*)&Bs[0], Wt + (size_t)nBase * K, 32, 64, (unsigned)K);
    }
    int cur = 0;
    for (int k0 = 0; k0 < K; k0 += 32, cur ^= 1) {
        if (wave == 0) __builtin_amdgcn_s_wait_tensorcnt(0);   // current tiles landed
        __syncthreads();                                        // release all waves
        if (wave == 0 && (k0 + 32) < K) {                       // prefetch next tiles
            int nxt = cur ^ 1;
            tdm_load_2d((unsigned)(size_t)(void*)&As[nxt * 4096],
                        A + (size_t)mBase * K + (k0 + 32), 32, 128, (unsigned)K);
            tdm_load_2d((unsigned)(size_t)(void*)&Bs[nxt * 2048],
                        Wt + (size_t)nBase * K + (k0 + 32), 32, 64, (unsigned)K);
        }
        const int ab = cur * 4096, bb = cur * 2048;
#else
    for (int k0 = 0; k0 < K; k0 += 32) {
        __syncthreads();
        #pragma unroll
        for (int i = 0; i < 2; ++i) {
            int ch = t + i * 256;
            int row = ch >> 2, c8 = ch & 3;
            *(uint4*)&As[row * 32 + c8 * 8] =
                *(const uint4*)(A + (size_t)(mBase + row) * K + k0 + c8 * 8);
        }
        {
            int row = t >> 2, c8 = t & 3;
            *(uint4*)&Bs[row * 32 + c8 * 8] =
                *(const uint4*)(Wt + (size_t)(nBase + row) * K + k0 + c8 * 8);
        }
        __syncthreads();
        const int ab = 0, bb = 0;
#endif
        // A fragment: lane L<16 holds row L, K {0..7,16..23}; L>=16: K {8..15,24..31}
        frag16 a0, a1, b0, b1;
        {
            int r = wm * 32 + mrow;
            a0.u[0] = *(const uint4*)&As[ab + r * 32 + kA];
            a0.u[1] = *(const uint4*)&As[ab + r * 32 + kA + 16];
            a1.u[0] = *(const uint4*)&As[ab + (r + 16) * 32 + kA];
            a1.u[1] = *(const uint4*)&As[ab + (r + 16) * 32 + kA + 16];
        }
        {
            int r = wn * 32 + mrow;
            b0.u[0] = *(const uint4*)&Bs[bb + r * 32 + kB];
            b0.u[1] = *(const uint4*)&Bs[bb + r * 32 + kB + 16];
            b1.u[0] = *(const uint4*)&Bs[bb + (r + 16) * 32 + kB];
            b1.u[1] = *(const uint4*)&Bs[bb + (r + 16) * 32 + kB + 16];
        }
        acc00 = __builtin_amdgcn_wmma_f32_16x16x32_f16(false, a0.h, false, b0.h, (short)0, acc00, false, false);
        acc01 = __builtin_amdgcn_wmma_f32_16x16x32_f16(false, a0.h, false, b1.h, (short)0, acc01, false, false);
        acc10 = __builtin_amdgcn_wmma_f32_16x16x32_f16(false, a1.h, false, b0.h, (short)0, acc10, false, false);
        acc11 = __builtin_amdgcn_wmma_f32_16x16x32_f16(false, a1.h, false, b1.h, (short)0, acc11, false, false);
    }

    // Epilogue: bias, fp16 store, per-column partial sums into LDS, then global.
    const int rofs = (lane < 16) ? 0 : 8;
    v8f accs[2][2] = { {acc00, acc01}, {acc10, acc11} };
    #pragma unroll
    for (int ni = 0; ni < 2; ++ni) {
        int n = nBase + wn * 32 + ni * 16 + mrow;
        float bn = bias[n];
        float ps = 0.0f, pq = 0.0f;
        #pragma unroll
        for (int mi = 0; mi < 2; ++mi) {
            #pragma unroll
            for (int r = 0; r < 8; ++r) {
                int row = mBase + wm * 32 + mi * 16 + rofs + r;
                float y = accs[mi][ni][r] + bn;
                Y[(size_t)row * N + n] = (_Float16)y;
                ps += y; pq += y * y;
            }
        }
        atomicAdd(&s_sum[n - nBase], ps);   // ds_add_f32
        atomicAdd(&s_sq[n - nBase], pq);
    }
    __syncthreads();
    if (t < 64) {
        atomicAdd(&colsum[nBase + t], s_sum[t]);
        atomicAdd(&colsumsq[nBase + t], s_sq[t]);
    }
}

// ------------------------------ batchnorm finalize / apply ------------------
__global__ void k_finalize_stats(const float* cs, const float* cq,
                                 const float* gamma, const float* beta,
                                 float* scale, float* shift, int Ncols, float invM) {
    int t = threadIdx.x;
    if (t < Ncols) {
        float m = cs[t] * invM;
        float v = cq[t] * invM - m * m;
        float sc = gamma[t] * rsqrtf(v + EPS_BN);
        scale[t] = sc;
        shift[t] = beta[t] - m * sc;
    }
}

__global__ void k_bn_relu(const _Float16* __restrict__ Y, _Float16* __restrict__ X,
                          const float* __restrict__ scale, const float* __restrict__ shift,
                          long long total, int Ncols) {
    long long i8 = ((long long)blockIdx.x * 256 + threadIdx.x) * 8;
    if (i8 >= total) return;
    h8pack in, outv;
    in.u = *(const uint4*)(Y + i8);
    int c0 = (int)(i8 % Ncols);
    #pragma unroll
    for (int k = 0; k < 8; ++k) {
        float y = (float)in.h[k];
        outv.h[k] = (_Float16)fmaxf(y * scale[c0 + k] + shift[c0 + k], 0.0f);
    }
    *(uint4*)(X + i8) = outv.u;
}

__global__ void k_bn_relu_max(const _Float16* __restrict__ Y, _Float16* __restrict__ P,
                              const float* __restrict__ scale, const float* __restrict__ shift,
                              int BS, int ns, int Ncols) {
    int idx = blockIdx.x * 256 + threadIdx.x;
    if (idx >= BS * Ncols) return;
    int bs = idx / Ncols, c = idx - bs * Ncols;
    float sc = scale[c], sh = shift[c], m = 0.0f;
    const _Float16* col = Y + (size_t)bs * ns * Ncols + c;
    for (int r = 0; r < ns; ++r) {
        if ((r & 7) == 0 && (r + 8) < ns)
            __builtin_prefetch((const void*)(col + (size_t)(r + 8) * Ncols), 0, 0);
        float y = (float)col[(size_t)r * Ncols];
        m = fmaxf(m, fmaxf(y * sc + sh, 0.0f));
    }
    P[idx] = (_Float16)m;
}

__global__ void k_bn_relu_max_sa3(const _Float16* __restrict__ Y,
                                  const float* __restrict__ scale, const float* __restrict__ shift,
                                  float* __restrict__ feat, float* __restrict__ outTail) {
    int idx = blockIdx.x * 256 + threadIdx.x;
    if (idx >= 16 * 512) return;
    int b = idx >> 9, c = idx & 511;
    float sc = scale[c], sh = shift[c], m = 0.0f;
    for (int r = 0; r < 512; ++r) {
        float y = (float)Y[((size_t)(b * 512 + r)) * 512 + c];
        m = fmaxf(m, fmaxf(y * sc + sh, 0.0f));
    }
    feat[idx] = m;
    outTail[idx] = m;
}

// ------------------------------ FC head (tiny) ------------------------------
__global__ void k_fc_head(const float* __restrict__ feat,                       // [16][512]
                          const float* W1, const float* b1, const float* be1, const float* g1,
                          const float* W2, const float* b2, const float* be2, const float* g2,
                          const float* W3, const float* b3,
                          float* __restrict__ logits) {                         // [16][40]
    __shared__ float h1[16 * 256];
    __shared__ float h2[16 * 128];
    int t = threadIdx.x;
    if (t < 256) {
        float y[16];
        for (int r = 0; r < 16; ++r) {
            float s = b1[t];
            for (int k = 0; k < 512; ++k) s += feat[r * 512 + k] * W1[(size_t)k * 256 + t];
            y[r] = s;
        }
        float m = 0.0f; for (int r = 0; r < 16; ++r) m += y[r]; m *= (1.0f / 16.0f);
        float v = 0.0f; for (int r = 0; r < 16; ++r) { float d = y[r] - m; v += d * d; } v *= (1.0f / 16.0f);
        float sc = g1[t] * rsqrtf(v + EPS_BN), sh = be1[t] - m * sc;
        for (int r = 0; r < 16; ++r) h1[r * 256 + t] = fmaxf(y[r] * sc + sh, 0.0f);
    }
    __syncthreads();
    if (t < 128) {
        float y[16];
        for (int r = 0; r < 16; ++r) {
            float s = b2[t];
            for (int k = 0; k < 256; ++k) s += h1[r * 256 + k] * W2[(size_t)k * 128 + t];
            y[r] = s;
        }
        float m = 0.0f; for (int r = 0; r < 16; ++r) m += y[r]; m *= (1.0f / 16.0f);
        float v = 0.0f; for (int r = 0; r < 16; ++r) { float d = y[r] - m; v += d * d; } v *= (1.0f / 16.0f);
        float sc = g2[t] * rsqrtf(v + EPS_BN), sh = be2[t] - m * sc;
        for (int r = 0; r < 16; ++r) h2[r * 128 + t] = fmaxf(y[r] * sc + sh, 0.0f);
    }
    __syncthreads();
    if (t < 40) {
        for (int r = 0; r < 16; ++r) {
            float s = b3[t];
            for (int k = 0; k < 128; ++k) s += h2[r * 128 + k] * W3[(size_t)k * 40 + t];
            logits[r * 40 + t] = s;
        }
    }
}

// ---------------------------------------------------------------------------
// host-side orchestration
// ---------------------------------------------------------------------------
extern "C" void kernel_launch(void* const* d_in, const int* in_sizes, int n_in,
                              void* d_out, int out_size, void* d_ws, size_t ws_size,
                              hipStream_t stream) {
    (void)out_size; (void)ws_size;

    // ---- locate xyz (unique elem count 98304); params follow pytree order ----
    int xi = -1;
    for (int i = 0; i < n_in; ++i) if (in_sizes[i] == 16 * 3 * 2048) { xi = i; break; }
    if (xi < 0) xi = n_in - 1;
    auto P = [&](int p) -> const float* {
        int idx = (xi == 0) ? p + 1 : p;
        return (const float*)d_in[idx];
    };
    // pytree flat order: fc1{W,b,beta,gamma} fc2{..} fc3{W,b} sa1[3x{W,b,beta,gamma}] sa2 sa3
    const float* xyz = (const float*)d_in[xi];
    float* out = (float*)d_out;

    // ---- workspace carve-out (256B-aligned static offsets) ----
    char* ws = (char*)d_ws;
    const size_t OFF_XYZT  = 0;                         // 16*2048*3 f32
    const size_t OFF_NXYZ1 = OFF_XYZT  + 393216;        // 16*1024*3 f32
    const size_t OFF_NXYZ2 = OFF_NXYZ1 + 196608;        // 16*512*3 f32
    const size_t OFF_L1P   = OFF_NXYZ2 + 98304;         // 16*1024*128 fp16
    const size_t OFF_L2P   = OFF_L1P   + 4194304;       // 16*512*256 fp16
    const size_t OFF_FEAT  = OFF_L2P   + 4194304;       // 16*512 f32
    const size_t OFF_STAT  = OFF_FEAT  + 32768;         // 4x512 f32
    const size_t OFF_WT    = OFF_STAT  + 8192;          // packed fp16 weights
    const size_t OFF_X     = OFF_WT    + 1048576;       // 524288*160 fp16
    const size_t OFF_Y     = OFF_X     + 167772160;     // 524288*256 fp16

    float*    xyzT  = (float*)(ws + OFF_XYZT);
    float*    nxyz1 = (float*)(ws + OFF_NXYZ1);
    float*    nxyz2 = (float*)(ws + OFF_NXYZ2);
    _Float16* l1p   = (_Float16*)(ws + OFF_L1P);
    _Float16* l2p   = (_Float16*)(ws + OFF_L2P);
    float*    feat  = (float*)(ws + OFF_FEAT);
    float*    cs    = (float*)(ws + OFF_STAT);
    float*    cq    = cs + 512;
    float*    scl   = cs + 1024;
    float*    shf   = cs + 1536;
    _Float16* WT    = (_Float16*)(ws + OFF_WT);
    _Float16* X     = (_Float16*)(ws + OFF_X);
    _Float16* Y     = (_Float16*)(ws + OFF_Y);

    // packed weight sub-offsets (in halves)
    _Float16* wt10 = WT + 0;        // 64  x 32
    _Float16* wt11 = WT + 2048;     // 64  x 64
    _Float16* wt12 = WT + 6144;     // 128 x 64
    _Float16* wt20 = WT + 14336;    // 128 x 160
    _Float16* wt21 = WT + 34816;    // 128 x 128
    _Float16* wt22 = WT + 51200;    // 256 x 128
    _Float16* wt30 = WT + 83968;    // 256 x 288
    _Float16* wt31 = WT + 157696;   // 256 x 256
    _Float16* wt32 = WT + 223232;   // 512 x 256

    auto packw = [&](const float* W, _Float16* Wt, int cin, int cout, int kpad) {
        int n = cout * kpad;
        k_pack_w<<<(n + 255) / 256, 256, 0, stream>>>(W, Wt, cin, cout, kpad);
    };
    packw(P(10), wt10,   3,  64,  32);
    packw(P(14), wt11,  64,  64,  64);
    packw(P(18), wt12,  64, 128,  64);
    packw(P(22), wt20, 131, 128, 160);
    packw(P(26), wt21, 128, 128, 128);
    packw(P(30), wt22, 128, 256, 128);
    packw(P(34), wt30, 259, 256, 288);
    packw(P(38), wt31, 256, 256, 256);
    packw(P(42), wt32, 256, 512, 256);

    auto gemm_layer = [&](const _Float16* Ap, const _Float16* Wtp,
                          const float* bias, const float* beta, const float* gamma,
                          int M, int Ncols, int K) {
        k_zero_stats<<<1, 512, 0, stream>>>(cs, cq);
        dim3 g(Ncols / 64, M / 128);
        k_wmma_gemm_bn<<<g, 256, 0, stream>>>(Ap, Wtp, bias, Y, cs, cq, M, Ncols, K);
        k_finalize_stats<<<1, 512, 0, stream>>>(cs, cq, gamma, beta, scl, shf,
                                                Ncols, 1.0f / (float)M);
    };
    auto bnrelu = [&](int M, int Ncols) {
        long long total = (long long)M * Ncols;
        k_bn_relu<<<(unsigned)(total / 8 / 256), 256, 0, stream>>>(Y, X, scl, shf, total, Ncols);
    };

    const int R1 = 16 * 1024 * 32;   // 524288 rows (SA1 and SA2)
    const int R3 = 16 * 512;         // 8192 rows (SA3)

    // ---- stage 0: transpose ----
    k_transpose_xyz<<<(16 * 2048 * 3 + 255) / 256, 256, 0, stream>>>(xyz, xyzT);

    // ---- SA1 ----
    k_fps<<<16, 1024, 0, stream>>>(xyzT, nxyz1, 2048, 1024);
    k_group_sa1<<<16 * 1024, 32, 0, stream>>>(xyzT, nxyz1, X);
    gemm_layer(X, wt10, P(11), P(12), P(13), R1,  64,  32);  bnrelu(R1, 64);
    gemm_layer(X, wt11, P(15), P(16), P(17), R1,  64,  64);  bnrelu(R1, 64);
    gemm_layer(X, wt12, P(19), P(20), P(21), R1, 128,  64);
    k_bn_relu_max<<<(16384 * 128) / 256, 256, 0, stream>>>(Y, l1p, scl, shf, 16384, 32, 128);

    // ---- SA2 ----
    k_fps<<<16, 1024, 0, stream>>>(nxyz1, nxyz2, 1024, 512);
    k_group_sa2<<<16 * 512, 64, 0, stream>>>(nxyz1, nxyz2, l1p, X);
    gemm_layer(X, wt20, P(23), P(24), P(25), R1, 128, 160);  bnrelu(R1, 128);
    gemm_layer(X, wt21, P(27), P(28), P(29), R1, 128, 128);  bnrelu(R1, 128);
    gemm_layer(X, wt22, P(31), P(32), P(33), R1, 256, 128);
    k_bn_relu_max<<<(8192 * 256) / 256, 256, 0, stream>>>(Y, l2p, scl, shf, 8192, 64, 256);

    // ---- SA3 (group all) ----
    k_group_sa3<<<(8192 * 288) / 256, 256, 0, stream>>>(nxyz2, l2p, X);
    gemm_layer(X, wt30, P(35), P(36), P(37), R3, 256, 288);  bnrelu(R3, 256);
    gemm_layer(X, wt31, P(39), P(40), P(41), R3, 256, 256);  bnrelu(R3, 256);
    gemm_layer(X, wt32, P(43), P(44), P(45), R3, 512, 256);
    k_bn_relu_max_sa3<<<(16 * 512) / 256, 256, 0, stream>>>(Y, scl, shf, feat, out + 640);

    // ---- FC head ----
    k_fc_head<<<1, 256, 0, stream>>>(feat,
                                     P(0), P(1), P(2), P(3),
                                     P(4), P(5), P(6), P(7),
                                     P(8), P(9),
                                     out);
}
